// HashNetwork_46892452938445
// MI455X (gfx1250) — matrix operand order repaired and checked by reading, not compiled, not run on hardware
//
#include <hip/hip_runtime.h>

typedef __attribute__((ext_vector_type(16))) _Float16 v16h;
typedef __attribute__((ext_vector_type(8)))  float    v8f;

#define NLVL 16
#define TMASK ((1u << 19) - 1u)
#define TSTRIDE (1u << 19)
#define DHID 64
#define ENC_SCALE 4096.0f
#define ENC_INV_SCALE (1.0f / 4096.0f)
#define WAVES_PER_BLOCK 8
#define PTS_PER_GROUP 128

// RES[l] = floor(16 * (4096/16)^(l/15)), computed in float64 (matches numpy).
// Levels 0..4 index directly ((RES+1)^3 <= 2^19); levels 5..15 hash.
__device__ __constant__ unsigned kRes[NLVL] = {
    16, 23, 33, 48, 70, 101, 147, 212, 307, 445, 645, 933, 1351, 1955, 2830, 4096};

union PackU { _Float16 h[2]; unsigned u; };
union FragU { v16h v; unsigned u[8]; };

__global__ __launch_bounds__(256)
void hashnet_fused_kernel(const float* __restrict__ r,
                          const float* __restrict__ phi,
                          const float* __restrict__ tab_m, const float* __restrict__ w1_m,
                          const float* __restrict__ b1_m,  const float* __restrict__ w2_m,
                          const float* __restrict__ b2_m,
                          const float* __restrict__ tab_p, const float* __restrict__ w1_p,
                          const float* __restrict__ b1_p,  const float* __restrict__ w2_p,
                          const float* __restrict__ b2_p,
                          float* __restrict__ out, int npts, int ngroups)
{
    // All LDS slices are PER-WAVE private: no block barriers needed anywhere
    // (per-wave in-order DS + compiler dscnt waits give intra-wave visibility).
    __shared__ unsigned encLds[WAVES_PER_BLOCK][2][16][17]; // [wave][net][point][kpair], padded
    __shared__ float    solLds[WAVES_PER_BLOCK][2][16];     // [wave][net][point]

    const int tid  = threadIdx.x;
    const int wave = tid >> 5;
    const int lane = tid & 31;
    const int half = lane >> 4;   // 0/1: net during encode; K-half during WMMA
    const int l16  = lane & 15;

    // ---- per-lane weight fragments for layer 1 (loaded once, reused per group) ----
    // B uses contiguous k (as A): lanes 0-15 hold K=0..15 (VGPR v -> K=2v,2v+1), lanes 16-31 hold K=16..31.
    v16h  bfrag[2][4];
    float b1v[2][4], w2v[2][4];
    #pragma unroll
    for (int net = 0; net < 2; ++net) {
        const float* w1 = net ? w1_p : w1_m;
        const float* b1 = net ? b1_p : b1_m;
        const float* w2 = net ? w2_p : w2_m;
        #pragma unroll
        for (int cb = 0; cb < 4; ++cb) {
            const int col = cb * 16 + l16;
            FragU f;
            #pragma unroll
            for (int v = 0; v < 8; ++v) {
                const int k = 2 * v + 16 * half;
                PackU pk;
                pk.h[0] = (_Float16)w1[(k    ) * DHID + col];
                pk.h[1] = (_Float16)w1[(k + 1) * DHID + col];
                f.u[v] = pk.u;
            }
            bfrag[net][cb] = f.v;
            b1v[net][cb] = b1[col];
            w2v[net][cb] = w2[col];
        }
    }
    const float b2mv = b2_m[0];
    const float b2pv = b2_p[0];

    for (int g = blockIdx.x; g < ngroups; g += gridDim.x) {
        const int tileBase = g * PTS_PER_GROUP + wave * 16;

        // ---- phase 1: hash-grid encoding; lane = (point l16, net half) ----
        {
            int p = tileBase + l16;
            if (p >= npts) p = npts - 1;
            const float* __restrict__ tab = half ? tab_p : tab_m;
            const float x = fminf(fmaxf((r[3 * p + 0] + 2.0f) * 0.25f, 0.0f), 1.0f);
            const float y = fminf(fmaxf((r[3 * p + 1] + 2.0f) * 0.25f, 0.0f), 1.0f);
            const float z = fminf(fmaxf((r[3 * p + 2] + 2.0f) * 0.25f, 0.0f), 1.0f);

            // levels 0..4: direct indexing, resolutions folded to immediates
            #pragma unroll
            for (int lev = 0; lev < 5; ++lev) {
                const unsigned res = kRes[lev];
                const float fres = (float)res;
                const float px = x * fres, py = y * fres, pz = z * fres;
                const float fx = floorf(px), fy = floorf(py), fz = floorf(pz);
                const float tx = px - fx,  ty = py - fy,  tz = pz - fz;
                const unsigned ix = (unsigned)fx, iy = (unsigned)fy, iz = (unsigned)fz;
                const unsigned stride  = res + 1u;
                const unsigned stride2 = stride * stride;
                float f0 = 0.0f, f1 = 0.0f;
                #pragma unroll
                for (int c = 0; c < 8; ++c) {
                    const unsigned dx = c & 1, dy = (c >> 1) & 1, dz = (c >> 2) & 1;
                    const unsigned idx =
                        ((ix + dx) + (iy + dy) * stride + (iz + dz) * stride2) & TMASK;
                    const float w = (dx ? tx : 1.0f - tx) *
                                    (dy ? ty : 1.0f - ty) *
                                    (dz ? tz : 1.0f - tz);
                    const float2 tv =
                        *(const float2*)(tab + (size_t)((unsigned)lev * TSTRIDE + idx) * 2u);
                    f0 = fmaf(w, tv.x, f0);
                    f1 = fmaf(w, tv.y, f1);
                }
                PackU pk;
                pk.h[0] = (_Float16)(f0 * ENC_SCALE);
                pk.h[1] = (_Float16)(f1 * ENC_SCALE);
                encLds[wave][half][l16][lev] = pk.u;  // level == K-pair index
            }

            // levels 5..15: hashed indexing
            for (int lev = 5; lev < NLVL; ++lev) {
                const unsigned res = kRes[lev];
                const float fres = (float)res;
                const float px = x * fres, py = y * fres, pz = z * fres;
                const float fx = floorf(px), fy = floorf(py), fz = floorf(pz);
                const float tx = px - fx,  ty = py - fy,  tz = pz - fz;
                const unsigned ix = (unsigned)fx, iy = (unsigned)fy, iz = (unsigned)fz;
                float f0 = 0.0f, f1 = 0.0f;
                #pragma unroll
                for (int c = 0; c < 8; ++c) {
                    const unsigned dx = c & 1, dy = (c >> 1) & 1, dz = (c >> 2) & 1;
                    const unsigned idx =
                        ((ix + dx) ^ ((iy + dy) * 2654435761u) ^ ((iz + dz) * 805459861u)) & TMASK;
                    const float w = (dx ? tx : 1.0f - tx) *
                                    (dy ? ty : 1.0f - ty) *
                                    (dz ? tz : 1.0f - tz);
                    const float2 tv =
                        *(const float2*)(tab + (size_t)((unsigned)lev * TSTRIDE + idx) * 2u);
                    f0 = fmaf(w, tv.x, f0);
                    f1 = fmaf(w, tv.y, f1);
                }
                PackU pk;
                pk.h[0] = (_Float16)(f0 * ENC_SCALE);
                pk.h[1] = (_Float16)(f1 * ENC_SCALE);
                encLds[wave][half][l16][lev] = pk.u;
            }
        }

        // ---- phase 2: layer1 via WMMA, layer2 via VALU + lane reduction ----
        // (same-wave producer/consumer: hardware keeps DS ops in order per wave)
        #pragma unroll
        for (int net = 0; net < 2; ++net) {
            // A uses contiguous k (as B): lanes 0-15 M=lane (K 0-15); lanes 16-31 M=lane-16 (K 16-31)
            FragU fa;
            #pragma unroll
            for (int v = 0; v < 8; ++v) {
                const int kpair = v + half * 8;
                fa.u[v] = encLds[wave][net][l16][kpair];
            }
            const v16h a = fa.v;

            float part[8] = {0.f, 0.f, 0.f, 0.f, 0.f, 0.f, 0.f, 0.f};
            #pragma unroll
            for (int cb = 0; cb < 4; ++cb) {
                v8f acc = {};
                acc = __builtin_amdgcn_wmma_f32_16x16x32_f16(
                    false, a, false, bfrag[net][cb], (short)0, acc, false, false);
                const float bb = b1v[net][cb];
                const float ww = w2v[net][cb];
                #pragma unroll
                for (int v = 0; v < 8; ++v) {
                    float h = fmaf(acc[v], ENC_INV_SCALE, bb); // un-scale + bias
                    h = fmaxf(h, 0.0f);                        // relu
                    part[v] = fmaf(h, ww, part[v]);            // layer-2 partial
                }
            }
            // reduce over the 16 lanes of each half (rows: low half M=v, high half M=v+8)
            #pragma unroll
            for (int m = 1; m < 16; m <<= 1) {
                #pragma unroll
                for (int v = 0; v < 8; ++v)
                    part[v] += __shfl_xor(part[v], m, 32);
            }
            if (l16 == 0) {
                #pragma unroll
                for (int v = 0; v < 8; ++v)
                    solLds[wave][net][half * 8 + v] = part[v];
            }
        }

        // ---- phase 3: select by sign(phi), add output bias, store ----
        if (half == 0) {
            const int p = tileBase + l16;
            if (p < npts) {
                const float sm = solLds[wave][0][l16] + b2mv;
                const float sp = solLds[wave][1][l16] + b2pv;
                out[p] = (phi[p] >= 0.0f) ? sp : sm;
            }
        }
    }
}

extern "C" void kernel_launch(void* const* d_in, const int* in_sizes, int n_in,
                              void* d_out, int out_size, void* d_ws, size_t ws_size,
                              hipStream_t stream) {
    const float* r     = (const float*)d_in[0];
    const float* phi   = (const float*)d_in[1];
    const float* tab_m = (const float*)d_in[2];
    const float* w1_m  = (const float*)d_in[3];
    const float* b1_m  = (const float*)d_in[4];
    const float* w2_m  = (const float*)d_in[5];
    const float* b2_m  = (const float*)d_in[6];
    const float* tab_p = (const float*)d_in[7];
    const float* w1_p  = (const float*)d_in[8];
    const float* b1_p  = (const float*)d_in[9];
    const float* w2_p  = (const float*)d_in[10];
    const float* b2_p  = (const float*)d_in[11];
    float* out = (float*)d_out;

    const int npts = in_sizes[0] / 3;
    const int ngroups = (npts + PTS_PER_GROUP - 1) / PTS_PER_GROUP;
    int blocks = ngroups < 512 ? ngroups : 512;
    if (blocks < 1) blocks = 1;

    hashnet_fused_kernel<<<blocks, 256, 0, stream>>>(
        r, phi, tab_m, w1_m, b1_m, w2_m, b2_m,
        tab_p, w1_p, b1_p, w2_p, b2_p, out, npts, ngroups);
}
